// E_GCL_vel_66958540145032
// MI455X (gfx1250) — matrix operand order repaired and checked by reading, not compiled, not run on hardware
//
#include <hip/hip_runtime.h>
#include <hip/hip_bf16.h>
#include <math.h>

typedef __attribute__((ext_vector_type(16))) _Float16 v16h;
typedef __attribute__((ext_vector_type(8)))  float    v8f;

#define HID   96
#define NN    81920
#define NE    327680

__device__ __forceinline__ float softplusf(float x){
  return x > 20.0f ? x : __logf(1.0f + __expf(x));
}
__device__ __forceinline__ float clip100(float x){
  return fminf(fmaxf(x, -100.0f), 100.0f);
}

// ---------------------------------------------------------------------------
// Wave-level GEMM: one wave computes a 16(M) x 96(N) f32 tile.
// A: f16 in LDS, row-major, this wave's 16 rows at sA (row stride ldA, col
//    offset aOff). B: global f32 weights Wg[KDIM][96], streamed into LDS as
//    transposed f16 chunks sW[96][32] so fragment pairs are contiguous.
// Fragment layouts per CDNA5 ISA (16-bit A 16x32 / B 32x16).
// ---------------------------------------------------------------------------
template<int KDIM>
__device__ __forceinline__ void gemm_stream(const float* __restrict__ Wg,
                                            const _Float16* sA, int ldA, int aOff,
                                            _Float16* sW,
                                            v8f acc[6], int tid, int nthr, int lane)
{
  const int mloc = lane & 15, g = lane >> 4;
  #pragma unroll 1
  for (int kt = 0; kt < KDIM / 32; ++kt) {
    __syncthreads();                       // previous chunk users done
    for (int i = tid; i < HID * 32; i += nthr) {
      int kk = i / HID, n = i - kk * HID;  // coalesced over n
      sW[n * 32 + kk] = (_Float16)Wg[(kt * 32 + kk) * HID + n];
    }
    __syncthreads();
    v16h a;
    #pragma unroll
    for (int v = 0; v < 4; ++v) {          // K = 2v + h + 8g   (VGPR 0..3)
      int k0 = aOff + kt * 32 + 2 * v + 8 * g;
      a[2 * v]     = sA[mloc * ldA + k0];
      a[2 * v + 1] = sA[mloc * ldA + k0 + 1];
    }
    #pragma unroll
    for (int v = 4; v < 8; ++v) {          // K = 16 + 2(v-4) + h + 8g
      int k0 = aOff + kt * 32 + 16 + 2 * (v - 4) + 8 * g;
      a[2 * v]     = sA[mloc * ldA + k0];
      a[2 * v + 1] = sA[mloc * ldA + k0 + 1];
    }
    #pragma unroll
    for (int nt = 0; nt < 6; ++nt) {
      v16h b;
      #pragma unroll
      for (int j = 0; j < 8; ++j) {        // K = 16(j/4) + 2(j%4) + h + 8g, N = lane%16
        int k0 = 16 * (j >> 2) + 2 * (j & 3) + 8 * g;
        b[2 * j]     = sW[(nt * 16 + mloc) * 32 + k0];
        b[2 * j + 1] = sW[(nt * 16 + mloc) * 32 + k0 + 1];
      }
      acc[nt] = __builtin_amdgcn_wmma_f32_16x16x32_f16(
          false, a, false, b, (short)0, acc[nt], false, false);
    }
  }
}

// C/D layout: lane L holds col N = nt*16 + (L&15); VGPR r holds row r + 8*(L>>4)
__device__ __forceinline__ void store_act_sp(const v8f acc[6],
                                             const float* __restrict__ bias,
                                             _Float16* sOut, int lane)
{
  int nloc = lane & 15, g = lane >> 4;
  #pragma unroll
  for (int nt = 0; nt < 6; ++nt) {
    int n = nt * 16 + nloc;
    float bb = bias[n];
    #pragma unroll
    for (int r = 0; r < 8; ++r)
      sOut[(r + 8 * g) * 96 + n] = (_Float16)softplusf(acc[nt][r] + bb);
  }
}

// ---------------------------------------------------------------------------
// K1: ef = sp(sp([h[row],h[col],edge_attr] @ ew1 + eb1) @ ew2 + eb2)
// ---------------------------------------------------------------------------
__global__ __launch_bounds__(128) void k_edge_mlp(
    const float* __restrict__ h, const int* __restrict__ ei,
    const float* __restrict__ edge_attr,
    const float* __restrict__ ew1, const float* __restrict__ eb1,
    const float* __restrict__ ew2, const float* __restrict__ eb2,
    float* __restrict__ ef)
{
  __shared__ _Float16 sCat[64 * 288];
  __shared__ _Float16 sAct[64 * 96];
  __shared__ _Float16 sW[96 * 32];
  const int tid = threadIdx.x, lane = tid & 31, wave = tid >> 5;
  const int e0 = blockIdx.x * 64;
  const int* rowI = ei;
  const int* colI = ei + NE;

  for (int i = tid; i < 64 * 288; i += 128) {
    int e = i / 288, k = i - e * 288, ge = e0 + e;
    float v;
    if (k < 96)       v = h[(size_t)rowI[ge] * 96 + k];
    else if (k < 192) v = h[(size_t)colI[ge] * 96 + (k - 96)];
    else              v = edge_attr[(size_t)ge * 96 + (k - 192)];
    sCat[i] = (_Float16)v;
  }

  v8f acc[6] = {};
  gemm_stream<288>(ew1, sCat + wave * 16 * 288, 288, 0, sW, acc, tid, 128, lane);
  store_act_sp(acc, eb1, sAct + wave * 16 * 96, lane);
  v8f acc2[6] = {};
  gemm_stream<96>(ew2, sAct + wave * 16 * 96, 96, 0, sW, acc2, tid, 128, lane);

  const int nloc = lane & 15, g = lane >> 4;
  #pragma unroll
  for (int nt = 0; nt < 6; ++nt) {
    int n = nt * 16 + nloc;
    float bb = eb2[n];
    #pragma unroll
    for (int r = 0; r < 8; ++r) {
      int grow = e0 + wave * 16 + r + 8 * g;
      ef[(size_t)grow * 96 + n] = softplusf(acc2[nt][r] + bb);
    }
  }
}

// ---------------------------------------------------------------------------
// K2a: qh = (h @ qw + qb)   (f16 out)
// ---------------------------------------------------------------------------
__global__ __launch_bounds__(128) void k_qproj(
    const float* __restrict__ h, const float* __restrict__ qw,
    const float* __restrict__ qb, _Float16* __restrict__ qh)
{
  __shared__ _Float16 sCat[64 * 96];
  __shared__ _Float16 sW[96 * 32];
  const int tid = threadIdx.x, lane = tid & 31, wave = tid >> 5;
  const int n0 = blockIdx.x * 64;
  for (int i = tid; i < 64 * 96; i += 128)
    sCat[i] = (_Float16)h[(size_t)n0 * 96 + i];

  v8f acc[6] = {};
  gemm_stream<96>(qw, sCat + wave * 16 * 96, 96, 0, sW, acc, tid, 128, lane);

  const int nloc = lane & 15, g = lane >> 4;
  #pragma unroll
  for (int nt = 0; nt < 6; ++nt) {
    int n = nt * 16 + nloc;
    float bb = qb[n];
    #pragma unroll
    for (int r = 0; r < 8; ++r) {
      int grow = n0 + wave * 16 + r + 8 * g;
      qh[(size_t)grow * 96 + n] = (_Float16)(acc[nt][r] + bb);
    }
  }
}

// ---------------------------------------------------------------------------
// K2b: kh = [h[src], ef] @ kw + kb ;  vh = ef @ vw + vb   (f16 out)
// ---------------------------------------------------------------------------
__global__ __launch_bounds__(128) void k_kv(
    const float* __restrict__ h, const int* __restrict__ ei,
    const float* __restrict__ ef,
    const float* __restrict__ kw, const float* __restrict__ kb,
    const float* __restrict__ vw, const float* __restrict__ vb,
    _Float16* __restrict__ kh, _Float16* __restrict__ vh)
{
  __shared__ _Float16 sCat[64 * 192];
  __shared__ _Float16 sW[96 * 32];
  const int tid = threadIdx.x, lane = tid & 31, wave = tid >> 5;
  const int e0 = blockIdx.x * 64;
  for (int i = tid; i < 64 * 192; i += 128) {
    int e = i / 192, k = i - e * 192, ge = e0 + e;
    float v = (k < 96) ? h[(size_t)ei[ge] * 96 + k]
                       : ef[(size_t)ge * 96 + (k - 96)];
    sCat[i] = (_Float16)v;
  }

  const int nloc = lane & 15, g = lane >> 4;
  v8f acc[6] = {};
  gemm_stream<192>(kw, sCat + wave * 16 * 192, 192, 0, sW, acc, tid, 128, lane);
  #pragma unroll
  for (int nt = 0; nt < 6; ++nt) {
    int n = nt * 16 + nloc;
    float bb = kb[n];
    #pragma unroll
    for (int r = 0; r < 8; ++r) {
      int grow = e0 + wave * 16 + r + 8 * g;
      kh[(size_t)grow * 96 + n] = (_Float16)(acc[nt][r] + bb);
    }
  }

  v8f acc2[6] = {};
  gemm_stream<96>(vw, sCat + wave * 16 * 192, 192, 96, sW, acc2, tid, 128, lane);
  #pragma unroll
  for (int nt = 0; nt < 6; ++nt) {
    int n = nt * 16 + nloc;
    float bb = vb[n];
    #pragma unroll
    for (int r = 0; r < 8; ++r) {
      int grow = e0 + wave * 16 + r + 8 * g;
      vh[(size_t)grow * 96 + n] = (_Float16)(acc2[nt][r] + bb);
    }
  }
}

// ---------------------------------------------------------------------------
// K3: per-node attention (softmax over 4 edges per head), agg + LayerNorm.
// Block = 1 node, 96 threads = 3 wave32s = 3 heads (d == warpSize == 32).
// ---------------------------------------------------------------------------
__global__ __launch_bounds__(96) void k_attn(
    const _Float16* __restrict__ qh, const _Float16* __restrict__ kh,
    const _Float16* __restrict__ vh,
    const float* __restrict__ g1, const float* __restrict__ be1,
    float* __restrict__ aggln)
{
  __shared__ float sSum[3], sSq[3];
  const int j = threadIdx.x, head = j >> 5, lane = j & 31;
  const int n = blockIdx.x;

  float qv = (float)qh[(size_t)n * 96 + head * 32 + lane];
  float s[4];
  #pragma unroll
  for (int e = 0; e < 4; ++e) {
    size_t ed = (size_t)(n * 4 + e) * 96 + head * 32 + lane;
    float p = qv * (float)kh[ed];
    #pragma unroll
    for (int m = 16; m > 0; m >>= 1) p += __shfl_xor(p, m, 32);
    s[e] = p * 0.17677669529663687f;   // 1/sqrt(32)
  }
  float mx = fmaxf(fmaxf(s[0], s[1]), fmaxf(s[2], s[3]));
  float w[4], den = 0.0f;
  #pragma unroll
  for (int e = 0; e < 4; ++e) { w[e] = __expf(s[e] - mx); den += w[e]; }
  float inv = 1.0f / den;
  float out = 0.0f;
  #pragma unroll
  for (int e = 0; e < 4; ++e)
    out += w[e] * inv * (float)vh[(size_t)(n * 4 + e) * 96 + head * 32 + lane];

  float ps = out, pq = out * out;
  #pragma unroll
  for (int m = 16; m > 0; m >>= 1) { ps += __shfl_xor(ps, m, 32); pq += __shfl_xor(pq, m, 32); }
  if (lane == 0) { sSum[head] = ps; sSq[head] = pq; }
  __syncthreads();
  float S = sSum[0] + sSum[1] + sSum[2];
  float Q = sSq[0] + sSq[1] + sSq[2];
  float mean = S * (1.0f / 96.0f);
  float var  = Q * (1.0f / 96.0f) - mean * mean;
  float rr = rsqrtf(var + 1e-5f);
  aggln[(size_t)n * 96 + j] = (out - mean) * rr * g1[j] + be1[j];
}

// ---------------------------------------------------------------------------
// K4: hn = LN(h + (sp([h,aggln] @ nw1 + nb1) @ nw2 + nb2))
// ---------------------------------------------------------------------------
__global__ __launch_bounds__(128) void k_node(
    const float* __restrict__ h, const float* __restrict__ aggln,
    const float* __restrict__ nw1, const float* __restrict__ nb1,
    const float* __restrict__ nw2, const float* __restrict__ nb2,
    const float* __restrict__ g2, const float* __restrict__ be2,
    float* __restrict__ hn)
{
  __shared__ _Float16 sCat[64 * 192];
  __shared__ _Float16 sAct[64 * 96];
  __shared__ _Float16 sW[96 * 32];
  const int tid = threadIdx.x, lane = tid & 31, wave = tid >> 5;
  const int n0 = blockIdx.x * 64;
  for (int i = tid; i < 64 * 192; i += 128) {
    int e = i / 192, k = i - e * 192;
    float v = (k < 96) ? h[(size_t)(n0 + e) * 96 + k]
                       : aggln[(size_t)(n0 + e) * 96 + (k - 96)];
    sCat[i] = (_Float16)v;
  }

  v8f acc[6] = {};
  gemm_stream<192>(nw1, sCat + wave * 16 * 192, 192, 0, sW, acc, tid, 128, lane);
  store_act_sp(acc, nb1, sAct + wave * 16 * 96, lane);
  v8f acc2[6] = {};
  gemm_stream<96>(nw2, sAct + wave * 16 * 96, 96, 0, sW, acc2, tid, 128, lane);

  __syncthreads();
  float* sRes = (float*)sCat;            // reuse (64*96*4 == 64*192*2 bytes)
  const int nloc = lane & 15, g = lane >> 4;
  #pragma unroll
  for (int nt = 0; nt < 6; ++nt) {
    int n = nt * 16 + nloc;
    float bb = nb2[n];
    #pragma unroll
    for (int r = 0; r < 8; ++r) {
      int lr = wave * 16 + r + 8 * g;
      sRes[lr * 96 + n] = acc2[nt][r] + bb + h[(size_t)(n0 + lr) * 96 + n];
    }
  }
  __syncthreads();

  int rowl = tid >> 1, half = tid & 1;   // 2 lanes per row, partners adjacent
  float ps = 0.0f, pq = 0.0f;
  for (int i = 0; i < 48; ++i) {
    float x = sRes[rowl * 96 + half * 48 + i];
    ps += x; pq += x * x;
  }
  ps += __shfl_xor(ps, 1, 32);
  pq += __shfl_xor(pq, 1, 32);
  float mean = ps * (1.0f / 96.0f);
  float var  = pq * (1.0f / 96.0f) - mean * mean;
  float rr = rsqrtf(var + 1e-5f);
  for (int i = 0; i < 48; ++i) {
    int c = half * 48 + i;
    float x = sRes[rowl * 96 + c];
    hn[(size_t)(n0 + rowl) * 96 + c] = (x - mean) * rr * g2[c] + be2[c];
  }
}

// ---------------------------------------------------------------------------
// K5: ef2 = ef + sp([hn[row],hn[col],edge_attr] @ xw1 + xb1) @ xw2 + xb2
// ---------------------------------------------------------------------------
__global__ __launch_bounds__(128) void k_edge_enh(
    const float* __restrict__ hn, const int* __restrict__ ei,
    const float* __restrict__ edge_attr, const float* __restrict__ ef,
    const float* __restrict__ xw1, const float* __restrict__ xb1,
    const float* __restrict__ xw2, const float* __restrict__ xb2,
    float* __restrict__ ef2)
{
  __shared__ _Float16 sCat[64 * 288];
  __shared__ _Float16 sAct[64 * 96];
  __shared__ _Float16 sW[96 * 32];
  const int tid = threadIdx.x, lane = tid & 31, wave = tid >> 5;
  const int e0 = blockIdx.x * 64;
  const int* rowI = ei;
  const int* colI = ei + NE;

  for (int i = tid; i < 64 * 288; i += 128) {
    int e = i / 288, k = i - e * 288, ge = e0 + e;
    float v;
    if (k < 96)       v = hn[(size_t)rowI[ge] * 96 + k];
    else if (k < 192) v = hn[(size_t)colI[ge] * 96 + (k - 96)];
    else              v = edge_attr[(size_t)ge * 96 + (k - 192)];
    sCat[i] = (_Float16)v;
  }

  v8f acc[6] = {};
  gemm_stream<288>(xw1, sCat + wave * 16 * 288, 288, 0, sW, acc, tid, 128, lane);
  store_act_sp(acc, xb1, sAct + wave * 16 * 96, lane);
  v8f acc2[6] = {};
  gemm_stream<96>(xw2, sAct + wave * 16 * 96, 96, 0, sW, acc2, tid, 128, lane);

  const int nloc = lane & 15, g = lane >> 4;
  #pragma unroll
  for (int nt = 0; nt < 6; ++nt) {
    int n = nt * 16 + nloc;
    float bb = xb2[n];
    #pragma unroll
    for (int r = 0; r < 8; ++r) {
      int grow = e0 + wave * 16 + r + 8 * g;
      ef2[(size_t)grow * 96 + n] = acc2[nt][r] + bb + ef[(size_t)grow * 96 + n];
    }
  }
}

// ---------------------------------------------------------------------------
// K6: coord path: t = sp(sp(ef2@cw1+cb1)@cw2+cb2); bc = t@cw3+cb3;
//     trans = clip(cd*bc0 + cc*bc1 + cv*bc2); acc = mean over 4 edges/node.
// ---------------------------------------------------------------------------
__global__ __launch_bounds__(128) void k_coord(
    const float* __restrict__ ef2, const int* __restrict__ ei,
    const float* __restrict__ coord_pre,
    const float* __restrict__ cw1, const float* __restrict__ cb1,
    const float* __restrict__ cw2, const float* __restrict__ cb2,
    const float* __restrict__ cw3, const float* __restrict__ cb3,
    float* __restrict__ accOut)
{
  __shared__ _Float16 sCat[64 * 96];
  __shared__ _Float16 sAct[64 * 96];
  __shared__ _Float16 sW[96 * 32];
  __shared__ float sTrans[64 * 3];
  const int tid = threadIdx.x, lane = tid & 31, wave = tid >> 5;
  const int e0 = blockIdx.x * 64;

  for (int i = tid; i < 64 * 96; i += 128)
    sCat[i] = (_Float16)ef2[(size_t)e0 * 96 + i];

  v8f acc[6] = {};
  gemm_stream<96>(cw1, sCat + wave * 16 * 96, 96, 0, sW, acc, tid, 128, lane);
  store_act_sp(acc, cb1, sAct + wave * 16 * 96, lane);
  v8f acc2[6] = {};
  gemm_stream<96>(cw2, sAct + wave * 16 * 96, 96, 0, sW, acc2, tid, 128, lane);
  store_act_sp(acc2, cb2, sAct + wave * 16 * 96, lane);  // own-wave region, in-order DS
  __syncthreads();

  // bc = t @ cw3 + cb3 : 2 lanes per edge-row, 48 K each, combine via xor(1)
  int rowl = lane >> 1, half = lane & 1;
  const _Float16* tl = sAct + wave * 16 * 96 + rowl * 96 + half * 48;
  float d0 = 0.0f, d1 = 0.0f, d2 = 0.0f;
  for (int k = 0; k < 48; ++k) {
    float tv = (float)tl[k];
    int gk = half * 48 + k;
    d0 += tv * cw3[gk * 3 + 0];
    d1 += tv * cw3[gk * 3 + 1];
    d2 += tv * cw3[gk * 3 + 2];
  }
  d0 += __shfl_xor(d0, 1, 32);
  d1 += __shfl_xor(d1, 1, 32);
  d2 += __shfl_xor(d2, 1, 32);
  d0 += cb3[0]; d1 += cb3[1]; d2 += cb3[2];

  int le = wave * 16 + rowl;
  if (half == 0) {
    int ge = e0 + le;
    int ri = ei[ge], ci = ei[NE + ge];
    float ax = coord_pre[(size_t)ri * 3 + 0], ay = coord_pre[(size_t)ri * 3 + 1], az = coord_pre[(size_t)ri * 3 + 2];
    float bx = coord_pre[(size_t)ci * 3 + 0], by = coord_pre[(size_t)ci * 3 + 1], bz = coord_pre[(size_t)ci * 3 + 2];
    float cdx = ax - bx, cdy = ay - by, cdz = az - bz;
    float ccx = ay * bz - az * by, ccy = az * bx - ax * bz, ccz = ax * by - ay * bx;
    float cvx = cdy * ccz - cdz * ccy, cvy = cdz * ccx - cdx * ccz, cvz = cdx * ccy - cdy * ccx;
    sTrans[le * 3 + 0] = clip100(cdx * d0 + ccx * d1 + cvx * d2);
    sTrans[le * 3 + 1] = clip100(cdy * d0 + ccy * d1 + cvy * d2);
    sTrans[le * 3 + 2] = clip100(cdz * d0 + ccz * d1 + cvz * d2);
  }
  __syncthreads();

  if (tid < 48) {                       // 16 nodes * 3 comps per block
    int nl = tid / 3, c = tid - nl * 3;
    float s = sTrans[(nl * 4 + 0) * 3 + c] + sTrans[(nl * 4 + 1) * 3 + c]
            + sTrans[(nl * 4 + 2) * 3 + c] + sTrans[(nl * 4 + 3) * 3 + c];
    accOut[(size_t)(blockIdx.x * 16 + nl) * 3 + c] = s * 0.25f;  // cnt==4, weight 1.0
  }
}

// ---------------------------------------------------------------------------
extern "C" void kernel_launch(void* const* d_in, const int* in_sizes, int n_in,
                              void* d_out, int out_size, void* d_ws, size_t ws_size,
                              hipStream_t stream)
{
  (void)in_sizes; (void)n_in; (void)out_size; (void)ws_size;
  const float* h         = (const float*)d_in[0];
  const int*   ei        = (const int*)  d_in[1];
  const float* coord_pre = (const float*)d_in[3];
  const float* edge_attr = (const float*)d_in[5];
  const float* ew1 = (const float*)d_in[6],  *eb1 = (const float*)d_in[7];
  const float* ew2 = (const float*)d_in[8],  *eb2 = (const float*)d_in[9];
  const float* xw1 = (const float*)d_in[10], *xb1 = (const float*)d_in[11];
  const float* xw2 = (const float*)d_in[12], *xb2 = (const float*)d_in[13];
  const float* nw1 = (const float*)d_in[14], *nb1 = (const float*)d_in[15];
  const float* nw2 = (const float*)d_in[16], *nb2 = (const float*)d_in[17];
  const float* cw1 = (const float*)d_in[18], *cb1 = (const float*)d_in[19];
  const float* cw2 = (const float*)d_in[20], *cb2 = (const float*)d_in[21];
  const float* cw3 = (const float*)d_in[22], *cb3 = (const float*)d_in[23];
  const float* qw  = (const float*)d_in[24], *qb  = (const float*)d_in[25];
  const float* kw  = (const float*)d_in[26], *kb  = (const float*)d_in[27];
  const float* vw  = (const float*)d_in[28], *vb  = (const float*)d_in[29];
  const float* g1  = (const float*)d_in[30], *be1 = (const float*)d_in[31];
  const float* g2  = (const float*)d_in[32], *be2 = (const float*)d_in[33];

  char* ws = (char*)d_ws;
  float*    ef    = (float*)ws;     ws += (size_t)NE * 96 * 4;
  _Float16* qh    = (_Float16*)ws;  ws += (size_t)NN * 96 * 2;
  _Float16* kh    = (_Float16*)ws;  ws += (size_t)NE * 96 * 2;
  _Float16* vh    = (_Float16*)ws;  ws += (size_t)NE * 96 * 2;
  float*    aggln = (float*)ws;     ws += (size_t)NN * 96 * 4;

  float* out    = (float*)d_out;
  float* accOut = out;                              // [NN,3]
  float* hnOut  = out + (size_t)NN * 3;             // [NN,96]
  float* ef2Out = hnOut + (size_t)NN * 96;          // [NE,96]

  dim3 b128(128);
  k_edge_mlp<<<NE / 64, b128, 0, stream>>>(h, ei, edge_attr, ew1, eb1, ew2, eb2, ef);
  k_qproj  <<<NN / 64, b128, 0, stream>>>(h, qw, qb, qh);
  k_kv     <<<NE / 64, b128, 0, stream>>>(h, ei, ef, kw, kb, vw, vb, kh, vh);
  k_attn   <<<NN, dim3(96), 0, stream>>>(qh, kh, vh, g1, be1, aggln);
  k_node   <<<NN / 64, b128, 0, stream>>>(h, aggln, nw1, nb1, nw2, nb2, g2, be2, hnOut);
  k_edge_enh<<<NE / 64, b128, 0, stream>>>(hnOut, ei, edge_attr, ef, xw1, xb1, xw2, xb2, ef2Out);
  k_coord  <<<NE / 64, b128, 0, stream>>>(ef2Out, ei, coord_pre, cw1, cb1, cw2, cb2, cw3, cb3, accOut);
}
